// Decoder_82764019794508
// MI455X (gfx1250) — matrix operand order repaired and checked by reading, not compiled, not run on hardware
//
#include <hip/hip_runtime.h>
#include <hip/hip_bf16.h>

typedef __attribute__((ext_vector_type(16))) _Float16 v16h;
typedef __attribute__((ext_vector_type(8)))  _Float16 v8h;
typedef __attribute__((ext_vector_type(8)))  float    v8f;

#define EPSV 1e-7f

// ---------------- geometry constants ----------------
#define B_   2
#define H2_  192          // fine grid
#define C2_  64
#define H4_  96
#define C4_  96
#define H32_ 24
#define C32_ 160
#define CMID 32
#define CCAT 96           // 3*32
#define HQ_  384
#define KDIM 864          // 9*96  (3x3 conv im2col K)
#define MLP_K 128         // 102 padded to 128
#define MLP_H 64

// ---------------- workspace layout (bytes, 256B aligned) ----------------
#define OFF_A4     0u                         // [2,96,96,32] f32   = 2359296
#define OFF_A32    2359296u                   // [2,24,24,32] f32   = 147456
#define OFF_WF16   2506752u                   // [864,96] f16       = 165888
#define OFF_W016   2672640u                   // [128,64] f16       = 16384
#define OFF_W116   2689024u                   // [64,64]  f16       = 8192
#define OFF_CAT16  2697216u                   // [2,192,192,96] f16 = 14155776
#define OFF_ASPP16 16852992u                  // [2,192,192,96] f16 = 14155776
// total = 31008768 bytes

// =====================================================================
// K0a: a4 = relu(feats4 @ w4 + b4)   [2,96,96,32] fp32
// =====================================================================
__global__ __launch_bounds__(256) void k_conv1x1_a4(
    const float* __restrict__ f4, const float* __restrict__ w4,
    const float* __restrict__ b4, float* __restrict__ a4) {
  int idx = blockIdx.x * 256 + threadIdx.x;
  if (idx >= B_ * H4_ * H4_ * CMID) return;
  int oc = idx & 31;
  int p  = idx >> 5;
  const float* in = f4 + (size_t)p * C4_;
  float acc = b4[oc];
  #pragma unroll 8
  for (int ic = 0; ic < C4_; ++ic) acc += in[ic] * w4[ic * CMID + oc];
  a4[idx] = acc > 0.f ? acc : 0.f;
}

// =====================================================================
// K0b: a32 = relu(feats32 @ w32 + b32)   [2,24,24,32] fp32
// =====================================================================
__global__ __launch_bounds__(256) void k_conv1x1_a32(
    const float* __restrict__ f32v, const float* __restrict__ w32,
    const float* __restrict__ b32, float* __restrict__ a32) {
  int idx = blockIdx.x * 256 + threadIdx.x;
  if (idx >= B_ * H32_ * H32_ * CMID) return;
  int oc = idx & 31;
  int p  = idx >> 5;
  const float* in = f32v + (size_t)p * C32_;
  float acc = b32[oc];
  #pragma unroll 8
  for (int ic = 0; ic < C32_; ++ic) acc += in[ic] * w32[ic * CMID + oc];
  a32[idx] = acc > 0.f ? acc : 0.f;
}

// =====================================================================
// K0c: pack weights to fp16.
//   wf (3,3,96,96) f32 -> wf16 [864,96]
//   mw0 (102,64)   f32 -> w016 [128,64]  (rows 102..127 zero)
//   mw1 (64,64)    f32 -> w116 [64,64]
// =====================================================================
__global__ __launch_bounds__(256) void k_pack_weights(
    const float* __restrict__ wf, const float* __restrict__ mw0,
    const float* __restrict__ mw1,
    _Float16* __restrict__ wf16, _Float16* __restrict__ w016,
    _Float16* __restrict__ w116) {
  int idx = blockIdx.x * 256 + threadIdx.x;
  const int N0 = KDIM * CCAT;           // 82944
  const int N1 = MLP_K * MLP_H;         // 8192
  const int N2 = MLP_H * MLP_H;         // 4096
  if (idx < N0) {
    wf16[idx] = (_Float16)wf[idx];
  } else if (idx < N0 + N1) {
    int j = idx - N0;
    int k = j >> 6, n = j & 63;
    w016[j] = (k < 102) ? (_Float16)mw0[k * 64 + n] : (_Float16)0.f;
  } else if (idx < N0 + N1 + N2) {
    int j = idx - N0 - N1;
    w116[j] = (_Float16)mw1[j];
  }
}

// =====================================================================
// K1: fused concat -> fp16 [2,192,192,96]
//   ch 0..31  : relu(feats2 @ w2 + b2)      (1x1 conv, 64 MAC)
//   ch 32..63 : bilinear(a4)  half-pixel, scale 1/2
//   ch 64..95 : bilinear(a32) half-pixel, scale 1/8
// =====================================================================
__global__ __launch_bounds__(256) void k_concat(
    const float* __restrict__ f2, const float* __restrict__ w2,
    const float* __restrict__ b2,
    const float* __restrict__ a4, const float* __restrict__ a32,
    _Float16* __restrict__ cat) {
  int idx = blockIdx.x * 256 + threadIdx.x;
  const int NTOT = B_ * H2_ * H2_ * CCAT;
  if (idx >= NTOT) return;
  int c = idx % CCAT;
  int p = idx / CCAT;
  int x = p % H2_;
  int y = (p / H2_) % H2_;
  int b = p / (H2_ * H2_);
  float val;
  if (c < 32) {
    const float* in = f2 + (size_t)p * C2_;
    float acc = b2[c];
    #pragma unroll 8
    for (int ic = 0; ic < C2_; ++ic) acc += in[ic] * w2[ic * CMID + c];
    val = acc > 0.f ? acc : 0.f;
  } else {
    int oc = c & 31;
    const float* src; int HS; float scale;
    if (c < 64) { src = a4;  HS = H4_;  scale = 0.5f;   }
    else        { src = a32; HS = H32_; scale = 0.125f; }
    float sy = (y + 0.5f) * scale - 0.5f;
    float sx = (x + 0.5f) * scale - 0.5f;
    float fy0 = floorf(sy), fx0 = floorf(sx);
    float wy = sy - fy0, wx = sx - fx0;
    int iy0 = (int)fy0, ix0 = (int)fx0;
    int iy1 = iy0 + 1, ix1 = ix0 + 1;
    iy0 = iy0 < 0 ? 0 : (iy0 > HS - 1 ? HS - 1 : iy0);
    iy1 = iy1 < 0 ? 0 : (iy1 > HS - 1 ? HS - 1 : iy1);
    ix0 = ix0 < 0 ? 0 : (ix0 > HS - 1 ? HS - 1 : ix0);
    ix1 = ix1 < 0 ? 0 : (ix1 > HS - 1 ? HS - 1 : ix1);
    size_t base = (size_t)b * HS * HS * CMID;
    float v00 = src[base + ((size_t)iy0 * HS + ix0) * CMID + oc];
    float v01 = src[base + ((size_t)iy0 * HS + ix1) * CMID + oc];
    float v10 = src[base + ((size_t)iy1 * HS + ix0) * CMID + oc];
    float v11 = src[base + ((size_t)iy1 * HS + ix1) * CMID + oc];
    val = (v00 * (1.f - wx) + v01 * wx) * (1.f - wy)
        + (v10 * (1.f - wx) + v11 * wx) * wy;
  }
  cat[idx] = (_Float16)val;
}

// =====================================================================
// K2: 3x3 conv 96->96 + relu as implicit GEMM via WMMA f16.
//   Per workgroup: 8x8 pixel tile -> M=64, N=96, K=864 (27 steps of 32).
//   256 threads = 8 waves; wave = (m_tile 0..3, n_half 0..1) -> 3 n-tiles.
// =====================================================================
__global__ __launch_bounds__(256) void k_conv3x3_wmma(
    const _Float16* __restrict__ src,   // [2,192,192,96] fp16
    const _Float16* __restrict__ wf16,  // [864,96] fp16 (im2col B matrix)
    const float* __restrict__ bfv,      // [96]
    _Float16* __restrict__ dst) {       // [2,192,192,96] fp16
  __shared__ __align__(16) _Float16 halo[10 * 10 * CCAT];  // 19200 B
  const int bx = blockIdx.x, by = blockIdx.y, b = blockIdx.z;
  const int tid = threadIdx.x;
  const int x0 = bx * 8 - 1, y0 = by * 8 - 1;

  // cooperative halo load, dword granularity, zero-padded borders
  for (int i = tid; i < 10 * 10 * (CCAT / 2); i += 256) {
    int hy = i / (10 * 48);
    int r  = i - hy * 10 * 48;
    int hx = r / 48;
    int cd = r - hx * 48;
    int gy = y0 + hy, gx = x0 + hx;
    unsigned v = 0u;
    if (gy >= 0 && gy < H2_ && gx >= 0 && gx < H2_)
      v = *(const unsigned*)(src + ((((size_t)b * H2_ + gy) * H2_ + gx) * CCAT + cd * 2));
    *(unsigned*)(&halo[(hy * 10 + hx) * CCAT + cd * 2]) = v;
  }
  __syncthreads();

  const int wave = tid >> 5, lane = tid & 31;
  const int mt = wave & 3;      // m tile (16 pixels)
  const int nh = wave >> 2;     // n half -> n-tiles nh*3 .. nh*3+2
  v8f acc0 = {}, acc1 = {}, acc2 = {};

  const int mrow = mt * 16 + (lane & 15);      // pixel 0..63 within tile
  const int py = mrow >> 3, px = mrow & 7;
  const int kb = (lane < 16) ? 0 : 8;          // A-layout K sub-base

  for (int ks = 0; ks < 27; ++ks) {
    const int tap = ks / 3;                    // 0..8 (3x3 tap)
    const int c0  = (ks - tap * 3) * 32;       // channel block within tap
    const int ky = tap / 3, kx = tap - ky * 3;

    // A fragment: 16-bit A 16x32 layout (K 0-7/16-23 low lanes, 8-15/24-31 high)
    const _Float16* ap = &halo[((py + ky) * 10 + (px + kx)) * CCAT + c0 + kb];
    v8h alo = *(const v8h*)ap;
    v8h ahi = *(const v8h*)(ap + 16);
    v16h a;
    #pragma unroll
    for (int j = 0; j < 8; ++j) { a[j] = alo[j]; a[8 + j] = ahi[j]; }

    const int krow = ks * 32 + kb;
    const _Float16* brow = wf16 + (size_t)krow * CCAT;
    if (ks + 1 < 27)  // prefetch next k-step's B row into caches
      __builtin_prefetch(brow + 32 * CCAT, 0, 0);

    #pragma unroll
    for (int nt = 0; nt < 3; ++nt) {
      const _Float16* bp = brow + (nh * 3 + nt) * 16 + (lane & 15);
      v16h bfrag;
      #pragma unroll
      for (int j = 0; j < 8; ++j) { bfrag[j] = bp[j * CCAT]; bfrag[8 + j] = bp[(16 + j) * CCAT]; }
      if (nt == 0)
        acc0 = __builtin_amdgcn_wmma_f32_16x16x32_f16(false, a, false, bfrag,
                                                      (short)0, acc0, false, false);
      else if (nt == 1)
        acc1 = __builtin_amdgcn_wmma_f32_16x16x32_f16(false, a, false, bfrag,
                                                      (short)0, acc1, false, false);
      else
        acc2 = __builtin_amdgcn_wmma_f32_16x16x32_f16(false, a, false, bfrag,
                                                      (short)0, acc2, false, false);
    }
  }

  // store: C layout -> VGPR r, lanes0-15: M=r,N=lane ; lanes16-31: M=8+r,N=lane-16
  const int nloc = lane & 15;
  const int mofs = (lane < 16) ? 0 : 8;
  #pragma unroll
  for (int nt = 0; nt < 3; ++nt) {
    const int n = (nh * 3 + nt) * 16 + nloc;
    const float bias = bfv[n];
    const v8f acc = (nt == 0) ? acc0 : (nt == 1) ? acc1 : acc2;
    #pragma unroll
    for (int r = 0; r < 8; ++r) {
      const int m = mt * 16 + mofs + r;
      const int oy = by * 8 + (m >> 3), ox = bx * 8 + (m & 7);
      float v = acc[r] + bias;
      v = v > 0.f ? v : 0.f;
      dst[(((size_t)b * H2_ + oy) * H2_ + ox) * CCAT + n] = (_Float16)v;
    }
  }
}

// =====================================================================
// K3: per-query gather + MLP (102->64->64->1) + 4-corner blend.
//   128 threads (4 waves), 64 query points per workgroup.
//   Layer1: M=64,N=64,K=128 (4 wmma k-steps); Layer2: K=64 (2 steps).
// =====================================================================
__global__ __launch_bounds__(128) void k_mlp_blend(
    const _Float16* __restrict__ aspp,   // [2,192,192,96] fp16
    const float* __restrict__ coords,    // [2,384,384,2]
    const float* __restrict__ cells,     // [2,384,384,2]
    const _Float16* __restrict__ w016,   // [128,64] fp16
    const float* __restrict__ mb0,       // [64]
    const _Float16* __restrict__ w116,   // [64,64] fp16
    const float* __restrict__ mb1,       // [64]
    const float* __restrict__ mw2,       // [64,1]
    const float* __restrict__ mb2,       // [1]
    float* __restrict__ out) {           // [2,384,384,1]
  __shared__ __align__(16) _Float16 Xs[64 * MLP_K];     // 16 KB
  __shared__ __align__(16) _Float16 W0s[MLP_K * MLP_H]; // 16 KB
  __shared__ __align__(16) _Float16 W1s[MLP_H * MLP_H]; //  8 KB
  __shared__ __align__(16) _Float16 Hs[64 * MLP_H];     //  8 KB
  __shared__ __align__(16) _Float16 H2s[64 * MLP_H];    //  8 KB
  __shared__ float b0s[64], b1s[64], w2s[64];
  __shared__ unsigned addrs[64];

  const int tid = threadIdx.x;
  // stage MLP weights/biases once
  for (int i = tid; i < MLP_K * MLP_H / 2; i += 128)
    ((unsigned*)W0s)[i] = ((const unsigned*)w016)[i];
  for (int i = tid; i < MLP_H * MLP_H / 2; i += 128)
    ((unsigned*)W1s)[i] = ((const unsigned*)w116)[i];
  if (tid < 64) { b0s[tid] = mb0[tid]; b1s[tid] = mb1[tid]; w2s[tid] = mw2[tid]; }

  // per-point setup (thread t<64 owns point t)
  const int pt = blockIdx.x * 64 + tid;      // valid for tid<64
  float cy = 0.f, cx = 0.f, rcy = 0.f, rcx = 0.f;
  int qb = 0;
  if (tid < 64) {
    int q  = pt;
    qb = q / (HQ_ * HQ_);
    int r = q - qb * HQ_ * HQ_;
    size_t cbase = ((size_t)qb * HQ_ * HQ_ + r) * 2;
    cy = coords[cbase + 0]; cx = coords[cbase + 1];
    rcy = cells[cbase + 0] * (float)H2_;
    rcx = cells[cbase + 1] * (float)H2_;
    // zero the padded K columns once (102..127 -> dwords 51..63)
    for (int d = 51; d < 64; ++d) ((unsigned*)Xs)[tid * 64 + d] = 0u;
  }

  const int wave = tid >> 5, lane = tid & 31;
  const int mt = wave;                       // m tile
  const int kb = (lane < 16) ? 0 : 8;
  const int nloc = lane & 15;
  const int mofs = (lane < 16) ? 0 : 8;
  const float rr = 1.f / (float)H2_;
  const float bias2 = mb2[0];

  float num = 0.f, den = 0.f;
  __syncthreads();

  for (int corner = 0; corner < 4; ++corner) {
    const float vy = (corner & 2) ? 1.f : -1.f;
    const float vx = (corner & 1) ? 1.f : -1.f;
    float area_op = 0.f;
    if (tid < 64) {
      // this corner's sample
      float ccy = fminf(fmaxf(cy + vy * rr + EPSV, -1.f + EPSV), 1.f - EPSV);
      float ccx = fminf(fmaxf(cx + vx * rr + EPSV, -1.f + EPSV), 1.f - EPSV);
      int iy = (int)roundf((ccy + 1.f) * 96.f - 0.5f);
      int ix = (int)roundf((ccx + 1.f) * 96.f - 0.5f);
      iy = iy < 0 ? 0 : (iy > H2_ - 1 ? H2_ - 1 : iy);
      ix = ix < 0 ? 0 : (ix > H2_ - 1 ? H2_ - 1 : ix);
      float csy = (iy + 0.5f) * (1.f / 96.f) - 1.f;
      float csx = (ix + 0.5f) * (1.f / 96.f) - 1.f;
      float rely = (cy - csy) * (float)H2_;
      float relx = (cx - csx) * (float)H2_;
      addrs[tid] = (unsigned)((((unsigned)qb * H2_ + iy) * H2_ + ix) * CCAT * 2);
      _Float16* xrow = &Xs[tid * MLP_K];
      xrow[96] = (_Float16)rely;  xrow[97] = (_Float16)relx;
      xrow[98] = (_Float16)cy;    xrow[99] = (_Float16)cx;
      xrow[100] = (_Float16)rcy;  xrow[101] = (_Float16)rcx;
      // opposite corner's area (pairs pred[i] with area[3-i])
      float ocy = fminf(fmaxf(cy - vy * rr + EPSV, -1.f + EPSV), 1.f - EPSV);
      float ocx = fminf(fmaxf(cx - vx * rr + EPSV, -1.f + EPSV), 1.f - EPSV);
      int jy = (int)roundf((ocy + 1.f) * 96.f - 0.5f);
      int jx = (int)roundf((ocx + 1.f) * 96.f - 0.5f);
      jy = jy < 0 ? 0 : (jy > H2_ - 1 ? H2_ - 1 : jy);
      jx = jx < 0 ? 0 : (jx > H2_ - 1 ? H2_ - 1 : jx);
      float osy = (jy + 0.5f) * (1.f / 96.f) - 1.f;
      float osx = (jx + 0.5f) * (1.f / 96.f) - 1.f;
      area_op = fabsf((cy - osy) * (float)H2_ * (cx - osx) * (float)H2_) + EPSV;
    }
    __syncthreads();
    // cooperative feature gather: 64 points x 48 dwords (96 fp16)
    for (int i = tid; i < 64 * 48; i += 128) {
      int p = i / 48, d = i - (i / 48) * 48;
      ((unsigned*)Xs)[p * 64 + d] =
          *(const unsigned*)((const char*)aspp + addrs[p] + d * 4);
    }
    __syncthreads();

    // ---- layer 1: X[64x128] @ W0[128x64] ----
    {
      v8f a0 = {}, a1 = {}, a2 = {}, a3 = {};
      const int m = mt * 16 + (lane & 15);
      #pragma unroll
      for (int ks = 0; ks < 4; ++ks) {
        const _Float16* ap = &Xs[m * MLP_K + ks * 32 + kb];
        v8h alo = *(const v8h*)ap;
        v8h ahi = *(const v8h*)(ap + 16);
        v16h a;
        #pragma unroll
        for (int j = 0; j < 8; ++j) { a[j] = alo[j]; a[8 + j] = ahi[j]; }
        const int krow = ks * 32 + kb;
        const _Float16* brow = &W0s[krow * MLP_H];
        #pragma unroll
        for (int nt = 0; nt < 4; ++nt) {
          const _Float16* bp = brow + nt * 16 + (lane & 15);
          v16h bf16;
          #pragma unroll
          for (int j = 0; j < 8; ++j) { bf16[j] = bp[j * MLP_H]; bf16[8 + j] = bp[(16 + j) * MLP_H]; }
          if (nt == 0) a0 = __builtin_amdgcn_wmma_f32_16x16x32_f16(false, a, false, bf16, (short)0, a0, false, false);
          else if (nt == 1) a1 = __builtin_amdgcn_wmma_f32_16x16x32_f16(false, a, false, bf16, (short)0, a1, false, false);
          else if (nt == 2) a2 = __builtin_amdgcn_wmma_f32_16x16x32_f16(false, a, false, bf16, (short)0, a2, false, false);
          else a3 = __builtin_amdgcn_wmma_f32_16x16x32_f16(false, a, false, bf16, (short)0, a3, false, false);
        }
      }
      #pragma unroll
      for (int nt = 0; nt < 4; ++nt) {
        const v8f acc = (nt == 0) ? a0 : (nt == 1) ? a1 : (nt == 2) ? a2 : a3;
        const int n = nt * 16 + nloc;
        const float bias = b0s[n];
        #pragma unroll
        for (int r = 0; r < 8; ++r) {
          float v = acc[r] + bias;
          v = v > 0.f ? v : 0.f;
          Hs[(mt * 16 + mofs + r) * MLP_H + n] = (_Float16)v;
        }
      }
    }
    __syncthreads();

    // ---- layer 2: H[64x64] @ W1[64x64] ----
    {
      v8f a0 = {}, a1 = {}, a2 = {}, a3 = {};
      const int m = mt * 16 + (lane & 15);
      #pragma unroll
      for (int ks = 0; ks < 2; ++ks) {
        const _Float16* ap = &Hs[m * MLP_H + ks * 32 + kb];
        v8h alo = *(const v8h*)ap;
        v8h ahi = *(const v8h*)(ap + 16);
        v16h a;
        #pragma unroll
        for (int j = 0; j < 8; ++j) { a[j] = alo[j]; a[8 + j] = ahi[j]; }
        const int krow = ks * 32 + kb;
        const _Float16* brow = &W1s[krow * MLP_H];
        #pragma unroll
        for (int nt = 0; nt < 4; ++nt) {
          const _Float16* bp = brow + nt * 16 + (lane & 15);
          v16h bf16;
          #pragma unroll
          for (int j = 0; j < 8; ++j) { bf16[j] = bp[j * MLP_H]; bf16[8 + j] = bp[(16 + j) * MLP_H]; }
          if (nt == 0) a0 = __builtin_amdgcn_wmma_f32_16x16x32_f16(false, a, false, bf16, (short)0, a0, false, false);
          else if (nt == 1) a1 = __builtin_amdgcn_wmma_f32_16x16x32_f16(false, a, false, bf16, (short)0, a1, false, false);
          else if (nt == 2) a2 = __builtin_amdgcn_wmma_f32_16x16x32_f16(false, a, false, bf16, (short)0, a2, false, false);
          else a3 = __builtin_amdgcn_wmma_f32_16x16x32_f16(false, a, false, bf16, (short)0, a3, false, false);
        }
      }
      #pragma unroll
      for (int nt = 0; nt < 4; ++nt) {
        const v8f acc = (nt == 0) ? a0 : (nt == 1) ? a1 : (nt == 2) ? a2 : a3;
        const int n = nt * 16 + nloc;
        const float bias = b1s[n];
        #pragma unroll
        for (int r = 0; r < 8; ++r) {
          float v = acc[r] + bias;
          v = v > 0.f ? v : 0.f;
          H2s[(mt * 16 + mofs + r) * MLP_H + n] = (_Float16)v;
        }
      }
    }
    __syncthreads();

    // ---- final projection + blend accumulation ----
    if (tid < 64) {
      float pred = bias2;
      const _Float16* hrow = &H2s[tid * MLP_H];
      #pragma unroll 8
      for (int k = 0; k < MLP_H; ++k) pred += (float)hrow[k] * w2s[k];
      num += pred * area_op;
      den += area_op;
    }
    __syncthreads();  // Xs/Hs reused next corner
  }

  if (tid < 64) out[pt] = num / den;
}

// =====================================================================
// launch
// =====================================================================
extern "C" void kernel_launch(void* const* d_in, const int* in_sizes, int n_in,
                              void* d_out, int out_size, void* d_ws, size_t ws_size,
                              hipStream_t stream) {
  (void)in_sizes; (void)n_in; (void)out_size; (void)ws_size;
  const float* feats2  = (const float*)d_in[0];
  const float* feats4  = (const float*)d_in[1];
  const float* feats32 = (const float*)d_in[2];
  const float* coords  = (const float*)d_in[3];
  const float* cells   = (const float*)d_in[4];
  const float* w2  = (const float*)d_in[5];
  const float* b2  = (const float*)d_in[6];
  const float* w4  = (const float*)d_in[7];
  const float* b4  = (const float*)d_in[8];
  const float* w32 = (const float*)d_in[9];
  const float* b32 = (const float*)d_in[10];
  const float* wf  = (const float*)d_in[11];
  const float* bfv = (const float*)d_in[12];
  const float* mw0 = (const float*)d_in[13];
  const float* mb0 = (const float*)d_in[14];
  const float* mw1 = (const float*)d_in[15];
  const float* mb1 = (const float*)d_in[16];
  const float* mw2 = (const float*)d_in[17];
  const float* mb2 = (const float*)d_in[18];
  float* out = (float*)d_out;

  char* ws = (char*)d_ws;
  float*     a4    = (float*)(ws + OFF_A4);
  float*     a32   = (float*)(ws + OFF_A32);
  _Float16*  wf16  = (_Float16*)(ws + OFF_WF16);
  _Float16*  w016  = (_Float16*)(ws + OFF_W016);
  _Float16*  w116  = (_Float16*)(ws + OFF_W116);
  _Float16*  cat16 = (_Float16*)(ws + OFF_CAT16);
  _Float16*  aspp  = (_Float16*)(ws + OFF_ASPP16);

  // 1) coarse 1x1 convs + weight packing (independent, stream-ordered)
  k_conv1x1_a4 <<<(B_*H4_*H4_*CMID + 255) / 256, 256, 0, stream>>>(feats4, w4, b4, a4);
  k_conv1x1_a32<<<(B_*H32_*H32_*CMID + 255) / 256, 256, 0, stream>>>(feats32, w32, b32, a32);
  k_pack_weights<<<(KDIM*CCAT + MLP_K*MLP_H + MLP_H*MLP_H + 255) / 256, 256, 0, stream>>>(
      wf, mw0, mw1, wf16, w016, w116);

  // 2) fused 1x1-conv + bilinear upsample + concat -> fp16
  k_concat<<<(B_*H2_*H2_*CCAT + 255) / 256, 256, 0, stream>>>(
      feats2, w2, b2, a4, a32, cat16);

  // 3) 3x3 conv (implicit GEMM, WMMA) -> aspp fp16
  k_conv3x3_wmma<<<dim3(H2_/8, H2_/8, B_), 256, 0, stream>>>(cat16, wf16, bfv, aspp);

  // 4) gather + MLP (WMMA) + 4-corner blend -> output
  k_mlp_blend<<<(B_*HQ_*HQ_) / 64, 128, 0, stream>>>(
      aspp, coords, cells, w016, mb0, w116, mb1, mw2, mb2, out);
}
